// GNNAutoencoder_80358838108850
// MI455X (gfx1250) — matrix operand order repaired and checked by reading, not compiled, not run on hardware
//
#include <hip/hip_runtime.h>
#include <hip/hip_bf16.h>

// ---------------------------------------------------------------------------
// GIN autoencoder on MI455X (gfx1250, wave32).
// GEMMs via v_wmma_f32_16x16x32_f16 (f16 in, f32 acc); edge aggregation via
// fp32 global atomics (h/agg both fit in the 192MB L2 -> atomics stay on-chip).
// ---------------------------------------------------------------------------

typedef __attribute__((ext_vector_type(16))) _Float16 v16h;
typedef __attribute__((ext_vector_type(8)))  _Float16 v8h;
typedef __attribute__((ext_vector_type(8)))  float    v8f;

static constexpr int N_ = 100000;   // nodes
static constexpr int E_ = 1600000;  // edges
static constexpr int G_ = 2048;     // graphs
static constexpr int H_ = 128;      // hidden
static constexpr int L_ = 3;        // GIN layers

// ---------------------------------------------------------------------------
// Weight convert + transpose: W[K][Nc] f32 row-major -> Wt[Nc][K] f16.
// Makes each lane's WMMA B-fragment (16 consecutive K) two contiguous B128s.
// ---------------------------------------------------------------------------
__global__ void cvt_transpose_kernel(const float* __restrict__ src,
                                     _Float16* __restrict__ dst,
                                     int K, int Nc) {
  int i = blockIdx.x * 256 + threadIdx.x;
  if (i >= K * Nc) return;
  int k = i / Nc;
  int n = i - k * Nc;
  dst[(size_t)n * K + k] = (_Float16)src[i];
}

__global__ void zero_kernel(float4* __restrict__ p, int n4) {
  int i = blockIdx.x * 256 + threadIdx.x;
  if (i < n4) p[i] = make_float4(0.f, 0.f, 0.f, 0.f);
}

// ---------------------------------------------------------------------------
// WMMA GEMM: C[M x (NT*16)] = act(A[M x (KT*32)] @ Bt^T + bias)
//   Bt is f16, stored transposed: [NT*16][KT*32].
//   One wave computes a 16-row strip x full output width.
//   rowIdx (optional) gathers A rows: A[rowIdx[row]] (decoder batch-gather).
// A-fragment layout (16-bit A 16x32, contiguous k like B):
//   lanes 0-15  (M=lane):    halves 0-15 -> K 0-15
//   lanes 16-31 (M=lane-16): halves 0-15 -> K 16-31
// B-fragment (16-bit B 32x16): N = lane&15; lanes<16 hold K 0-15, else K 16-31.
// C/D f32: N = lane&15, M = (lane>>4)*8 + reg.
// ---------------------------------------------------------------------------
template <int NT, int KT>
__global__ __launch_bounds__(256) void gemm_wmma_kernel(
    const float* __restrict__ A, const int* __restrict__ rowIdx, int lda,
    const _Float16* __restrict__ Bt, const float* __restrict__ bias,
    float* __restrict__ C, int ldc, int M, int relu) {
  constexpr int K = KT * 32;
  const int lane  = threadIdx.x & 31;
  const int wave  = threadIdx.x >> 5;
  const int strip = blockIdx.x * 8 + wave;
  if (strip * 16 >= M) return;  // wave-uniform: EXEC stays all-ones for WMMA

  const int mlo = lane & 15;
  const int hi  = lane >> 4;  // 0 or 1
  const int row = strip * 16 + mlo;
  const int ar  = rowIdx ? rowIdx[row] : row;
  const float* aptr = A + (size_t)ar * lda;

  v8f acc[NT];
#pragma unroll
  for (int nt = 0; nt < NT; ++nt) acc[nt] = (v8f){0.f, 0.f, 0.f, 0.f, 0.f, 0.f, 0.f, 0.f};

#pragma unroll
  for (int kt = 0; kt < KT; ++kt) {
    // ---- A fragment: one run of 16 consecutive f32, cvt to f16 ----
    const int kbase = kt * 32 + hi * 16;
    float4 q0 = *(const float4*)(aptr + kbase);
    float4 q1 = *(const float4*)(aptr + kbase + 4);
    float4 q2 = *(const float4*)(aptr + kbase + 8);
    float4 q3 = *(const float4*)(aptr + kbase + 12);
    v16h a;
    a[0]  = (_Float16)q0.x; a[1]  = (_Float16)q0.y;
    a[2]  = (_Float16)q0.z; a[3]  = (_Float16)q0.w;
    a[4]  = (_Float16)q1.x; a[5]  = (_Float16)q1.y;
    a[6]  = (_Float16)q1.z; a[7]  = (_Float16)q1.w;
    a[8]  = (_Float16)q2.x; a[9]  = (_Float16)q2.y;
    a[10] = (_Float16)q2.z; a[11] = (_Float16)q2.w;
    a[12] = (_Float16)q3.x; a[13] = (_Float16)q3.y;
    a[14] = (_Float16)q3.z; a[15] = (_Float16)q3.w;

    const int kb = kt * 32 + hi * 16;
#pragma unroll
    for (int nt = 0; nt < NT; ++nt) {
      const _Float16* bp = Bt + (size_t)(nt * 16 + mlo) * K + kb;
      v8h b0 = *(const v8h*)bp;
      v8h b1 = *(const v8h*)(bp + 8);
      v16h b;
#pragma unroll
      for (int i = 0; i < 8; ++i) { b[i] = b0[i]; b[i + 8] = b1[i]; }
      acc[nt] = __builtin_amdgcn_wmma_f32_16x16x32_f16(
          false, a, false, b, (short)0, acc[nt], false, false);
    }
  }

  // ---- epilogue: bias + optional relu, scatter by C/D layout ----
#pragma unroll
  for (int nt = 0; nt < NT; ++nt) {
    const int n   = nt * 16 + mlo;
    const float bb = bias ? bias[n] : 0.0f;
#pragma unroll
    for (int r = 0; r < 8; ++r) {
      float v = acc[nt][r] + bb;
      if (relu) v = fmaxf(v, 0.0f);
      C[(size_t)(strip * 16 + hi * 8 + r) * ldc + n] = v;
    }
  }
}

// ---------------------------------------------------------------------------
// Edge scatter-add: agg[dst[e]] += h[src[e]]   (one thread = edge x 4 ch)
// ---------------------------------------------------------------------------
__global__ void scatter_add_kernel(const float* __restrict__ h,
                                   const int* __restrict__ src,
                                   const int* __restrict__ dst,
                                   float* __restrict__ agg) {
  int gid = blockIdx.x * 256 + threadIdx.x;
  int e = gid >> 5;
  if (e >= E_) return;
  int c = (gid & 31) << 2;
  int s = src[e], d = dst[e];
  float4 v = *(const float4*)(h + (size_t)s * H_ + c);
  float* p = agg + (size_t)d * H_ + c;
  atomicAdd(p + 0, v.x);
  atomicAdd(p + 1, v.y);
  atomicAdd(p + 2, v.z);
  atomicAdd(p + 3, v.w);
}

// agg = (1+eps[l])*h + agg
__global__ void combine_kernel(const float* __restrict__ h, float* __restrict__ agg,
                               const float* __restrict__ eps, int l, int n4) {
  int i = blockIdx.x * 256 + threadIdx.x;
  if (i >= n4) return;
  float e = 1.0f + eps[l];
  float4 hv = ((const float4*)h)[i];
  float4 av = ((float4*)agg)[i];
  av.x = fmaf(e, hv.x, av.x);
  av.y = fmaf(e, hv.y, av.y);
  av.z = fmaf(e, hv.z, av.z);
  av.w = fmaf(e, hv.w, av.w);
  ((float4*)agg)[i] = av;
}

// pooled[batch[n]] += h[n]; counts[batch[n]] += 1
__global__ void pool_kernel(const float* __restrict__ h, const int* __restrict__ batch,
                            float* __restrict__ pooled, float* __restrict__ counts) {
  int gid = blockIdx.x * 256 + threadIdx.x;
  int n = gid >> 5;
  if (n >= N_) return;
  int c = (gid & 31) << 2;
  int g = batch[n];
  float4 v = *(const float4*)(h + (size_t)n * H_ + c);
  float* p = pooled + (size_t)g * H_ + c;
  atomicAdd(p + 0, v.x);
  atomicAdd(p + 1, v.y);
  atomicAdd(p + 2, v.z);
  atomicAdd(p + 3, v.w);
  if (c == 0) atomicAdd(&counts[g], 1.0f);
}

__global__ void divide_kernel(float* __restrict__ pooled, const float* __restrict__ counts) {
  int i = blockIdx.x * 256 + threadIdx.x;
  if (i >= G_ * H_) return;
  pooled[i] /= fmaxf(counts[i >> 7], 1.0f);
}

// ---------------------------------------------------------------------------
extern "C" void kernel_launch(void* const* d_in, const int* in_sizes, int n_in,
                              void* d_out, int out_size, void* d_ws, size_t ws_size,
                              hipStream_t stream) {
  const float* x     = (const float*)d_in[0];
  const int*   eidx  = (const int*)d_in[1];   // [2,E]: src row then dst row
  const int*   batch = (const int*)d_in[2];
  const float* W_in  = (const float*)d_in[3];
  const float* b_in  = (const float*)d_in[4];
  const float* Ws1   = (const float*)d_in[5];
  const float* bs1   = (const float*)d_in[6];
  const float* Ws2   = (const float*)d_in[7];
  const float* bs2   = (const float*)d_in[8];
  const float* eps   = (const float*)d_in[9];
  const float* W_out = (const float*)d_in[10];
  const float* b_out = (const float*)d_in[11];
  const float* Wd1   = (const float*)d_in[12];
  const float* bd1   = (const float*)d_in[13];
  const float* Wd2   = (const float*)d_in[14];
  const float* bd2   = (const float*)d_in[15];

  const int* esrc = eidx;
  const int* edst = eidx + E_;

  float* recon = (float*)d_out;                 // [N, 128]
  float* gemb  = recon + (size_t)N_ * H_;       // [G, 128]

  // workspace layout (all offsets multiples of 256B)
  float* h      = (float*)d_ws;                 // N*H
  float* agg    = h + (size_t)N_ * H_;          // N*H (also reused as N*64 mid)
  float* tmp    = agg + (size_t)N_ * H_;        // N*H
  float* pooled = tmp + (size_t)N_ * H_;        // G*H
  float* counts = pooled + (size_t)G_ * H_;     // G
  _Float16* wT_in  = (_Float16*)(counts + G_);  // 128*128
  _Float16* wT_s1  = wT_in + 16384;             // 3 * 128*128
  _Float16* wT_s2  = wT_s1 + 3 * 16384;         // 3 * 128*128
  _Float16* wT_out = wT_s2 + 3 * 16384;         // 128*128
  _Float16* wT_d1  = wT_out + 16384;            // 64 rows x 128 K
  _Float16* wT_d2  = wT_d1 + 8192;              // 128 rows x 64 K

  auto cvt = [&](const float* s, _Float16* d, int K, int Nc) {
    int total = K * Nc;
    cvt_transpose_kernel<<<(total + 255) / 256, 256, 0, stream>>>(s, d, K, Nc);
  };
  auto zero = [&](float* p, size_t nfloats) {
    int n4 = (int)(nfloats / 4);
    zero_kernel<<<(n4 + 255) / 256, 256, 0, stream>>>((float4*)p, n4);
  };
  auto gemm128 = [&](const float* A, const int* idx, int lda, const _Float16* Bt,
                     const float* bias, float* C, int ldc, int M, int relu) {
    int blocks = (M / 16 + 7) / 8;
    gemm_wmma_kernel<8, 4><<<blocks, 256, 0, stream>>>(A, idx, lda, Bt, bias, C, ldc, M, relu);
  };

  // 1) convert + transpose all weights to f16 once
  cvt(W_in, wT_in, 128, 128);
  for (int l = 0; l < L_; ++l) cvt(Ws1 + (size_t)l * 16384, wT_s1 + (size_t)l * 16384, 128, 128);
  for (int l = 0; l < L_; ++l) cvt(Ws2 + (size_t)l * 16384, wT_s2 + (size_t)l * 16384, 128, 128);
  cvt(W_out, wT_out, 128, 128);
  cvt(Wd1, wT_d1, 128, 64);   // [128,64] -> [64][128]
  cvt(Wd2, wT_d2, 64, 128);   // [64,128] -> [128][64]

  // 2) input projection: h = relu(x @ W_in + b_in)
  gemm128(x, nullptr, 128, wT_in, b_in, h, 128, N_, 1);

  // 3) GIN layers
  const int egrid = (E_ * 32) / 256;  // exact
  const int ngrid = (N_ * 32) / 256;  // exact
  for (int l = 0; l < L_; ++l) {
    zero(agg, (size_t)N_ * H_);
    scatter_add_kernel<<<egrid, 256, 0, stream>>>(h, esrc, edst, agg);
    combine_kernel<<<ngrid, 256, 0, stream>>>(h, agg, eps, l, N_ * H_ / 4);
    gemm128(agg, nullptr, 128, wT_s1 + (size_t)l * 16384, bs1 + (size_t)l * H_, tmp, 128, N_, 1);
    gemm128(tmp, nullptr, 128, wT_s2 + (size_t)l * 16384, bs2 + (size_t)l * H_, h, 128, N_, 1);
  }

  // 4) global mean pool + output projection -> graph embeddings (into d_out)
  zero(pooled, (size_t)G_ * H_ + G_);  // pooled and counts are contiguous
  pool_kernel<<<ngrid, 256, 0, stream>>>(h, batch, pooled, counts);
  divide_kernel<<<(G_ * H_ + 255) / 256, 256, 0, stream>>>(pooled, counts);
  gemm128(pooled, nullptr, 128, wT_out, b_out, gemb, 128, G_, 0);

  // 5) decoder: gather graph embedding per node (fused into A-load via batch),
  //    mid = relu(gemb[batch] @ Wd1 + bd1)  [N,64], reuse agg
  {
    int blocks = (N_ / 16 + 7) / 8;
    gemm_wmma_kernel<4, 4><<<blocks, 256, 0, stream>>>(
        gemb, batch, 128, wT_d1, bd1, agg, 64, N_, 1);
    //    recon = mid @ Wd2 + bd2  [N,128]
    gemm_wmma_kernel<8, 2><<<blocks, 256, 0, stream>>>(
        agg, nullptr, 64, wT_d2, bd2, recon, 128, N_, 0);
  }
}
